// GRUCellStack_33036888441278
// MI455X (gfx1250) — compile-verified, matrix operand order checked by reading, not run on hardware
//
#include <hip/hip_runtime.h>

// ---------------------------------------------------------------------------
// 2-layer GRU stack for MI455X (gfx1250, wave32, WMMA).
// B=8192, IN=L=2048.  ~825 GFLOP vs ~0.55GB traffic -> compute-bound; GEMMs
// run on v_wmma_f32_16x16x32_bf16 (fp32 accum).  LDS staging uses the CDNA5
// async-copy path (global_load_async_to_lds_b128 + s_wait_asynccnt) with
// double buffering so HBM traffic overlaps WMMA issue.
// ---------------------------------------------------------------------------

typedef __bf16 bf16_t;
typedef __attribute__((ext_vector_type(8)))  bf16_t v8bf;
typedef __attribute__((ext_vector_type(16))) bf16_t v16bf;
typedef __attribute__((ext_vector_type(8)))  float  v8f;

#define L_DIM 2048
#define B_DIM 8192
#define K_DIM 2048

// fp32 -> bf16 round-to-nearest-even
__device__ __forceinline__ unsigned short f2bf_rne(float f) {
    union { float f; unsigned int u; } c; c.f = f;
    unsigned int u = c.u;
    unsigned int r = u + 0x7FFFu + ((u >> 16) & 1u);
    return (unsigned short)(r >> 16);
}

__global__ void cvt_f32_bf16(const float* __restrict__ src,
                             unsigned short* __restrict__ dst, int n) {
    int i = blockIdx.x * blockDim.x + threadIdx.x;
    int stride = gridDim.x * blockDim.x;
    for (; i < n; i += stride) dst[i] = f2bf_rne(src[i]);
}

__global__ void cvt_f32_bf16_strided(const float* __restrict__ src,
                                     unsigned short* __restrict__ dst,
                                     int rows, int cols, int sstride, int soff) {
    int i = blockIdx.x * blockDim.x + threadIdx.x;
    int stride = gridDim.x * blockDim.x;
    int n = rows * cols;
    for (; i < n; i += stride) {
        int r = i / cols, c = i - r * cols;
        dst[i] = f2bf_rne(src[(long)r * sstride + soff + c]);
    }
}

// CDNA5 async copy: memory -> LDS directly, tracked by ASYNCcnt (not LOADcnt).
// %0 = LDS byte address (per-lane VGPR), %1 = 64-bit global address.
__device__ __forceinline__ void async_b128(unsigned lds, unsigned long long gaddr) {
    asm volatile("global_load_async_to_lds_b128 %0, %1, off"
                 :: "v"(lds), "v"(gaddr) : "memory");
}
// Generic LDS-aperture addresses carry the LDS byte offset in the low 32 bits
// (flat->LDS mapping truncates), so a 32-bit cast yields the DS address.
__device__ __forceinline__ unsigned lds_off(const void* p) {
    return (unsigned)(unsigned long long)p;
}

// One lane's 16-elem bf16 fragment from two 16B LDS chunks (K-split layout:
// lanes 0-15 hold K {0..7,16..23}, lanes 16-31 hold K {8..15,24..31}).
__device__ __forceinline__ v16bf frag16(const unsigned short* p0,
                                        const unsigned short* p1) {
    v8bf lo = *reinterpret_cast<const v8bf*>(p0);
    v8bf hi = *reinterpret_cast<const v8bf*>(p1);
    return __builtin_shufflevector(lo, hi,
        0,1,2,3,4,5,6,7,8,9,10,11,12,13,14,15);
}

__device__ __forceinline__ float sigmoidf(float x) {
    return 1.0f / (1.0f + __expf(-x));
}

#define WMMA_BF16(A, B, C) \
    __builtin_amdgcn_wmma_f32_16x16x32_bf16(false, (A), false, (B), (short)0, (C), false, false)

// Fused GRU layer: h_new = (1-z)*tanh(i_n + r*h_n) + z*h_prev.
// Block tile 128(batch) x 32(hidden); 8 waves, each owns a 32x16 tile
// (two 16x16 M-subtiles -> every weight fragment feeds two WMMAs).
// K loop step 32, double-buffered async LDS staging.
__launch_bounds__(256)
__global__ void gru_layer_wmma(const unsigned short* __restrict__ Xbf,  // [B,2048] bf16
                               const unsigned short* __restrict__ Hbf,  // [B,2048] bf16
                               const unsigned short* __restrict__ Wih,  // [6144,2048] bf16
                               const unsigned short* __restrict__ Whh,  // [6144,2048] bf16
                               const float* __restrict__ bih,           // [6144]
                               const float* __restrict__ bhh,           // [6144]
                               const float* __restrict__ state,         // [B,4096] fp32
                               int state_coff,
                               float* __restrict__ out,                 // [B,4096] fp32
                               int out_coff,
                               unsigned short* __restrict__ xnext)      // [B,2048] bf16 or null
{
    __shared__ unsigned short Xs[2][128][32];
    __shared__ unsigned short Hs[2][128][32];
    __shared__ unsigned short Ws[2][6][32][32];   // gates r,z,n x {W_ih, W_hh}

    const int t    = threadIdx.x;
    const int lane = t & 31;
    const int wave = t >> 5;
    const int wm   = wave & 3;    // 4 wave-rows  x 32 batch rows = 128
    const int wn   = wave >> 2;   // 2 wave-cols  x 16 hidden    = 32
    const int half = lane >> 4;
    const int l15  = lane & 15;

    const int m0 = blockIdx.y * 128;
    const int n0 = blockIdx.x * 32;

    v8f accr[2] = {}; v8f accz[2] = {}; v8f accin[2] = {}; v8f acchn[2] = {};

    // issue this thread's 7 async b128 copies for K-chunk kb into buffer buf
    auto stage = [&](int buf, int kb) {
        #pragma unroll
        for (int j = 0; j < 2; ++j) {                 // X,H: 128x32 -> 512 chunks each
            int c  = t + j * 256;                     // 0..511
            int r  = c >> 2;
            int c8 = (c & 3) * 8;
            async_b128(lds_off(&Xs[buf][r][c8]),
                       (unsigned long long)&Xbf[(long)(m0 + r) * K_DIM + kb + c8]);
            async_b128(lds_off(&Hs[buf][r][c8]),
                       (unsigned long long)&Hbf[(long)(m0 + r) * K_DIM + kb + c8]);
        }
        #pragma unroll
        for (int j = 0; j < 3; ++j) {                 // 6 weight tiles 32x32 -> 768 chunks
            int idx = t + j * 256;                    // 0..767
            int g   = idx >> 7;
            int r   = (idx >> 2) & 31;
            int c8  = (idx & 3) * 8;
            const unsigned short* src = (g < 3)
                ? &Wih[(long)(g * L_DIM + n0 + r) * K_DIM + kb + c8]
                : &Whh[(long)((g - 3) * L_DIM + n0 + r) * K_DIM + kb + c8];
            async_b128(lds_off(&Ws[buf][g][r][c8]), (unsigned long long)src);
        }
    };

    stage(0, 0);
    int cur = 0;
    for (int kb = 0; kb < K_DIM; kb += 32) {
        const int  nxt  = cur ^ 1;
        const bool more = (kb + 32) < K_DIM;          // uniform
        if (more) {
            stage(nxt, kb + 32);                      // 7 more in flight (14 total)
            // async loads retire in order: <=7 outstanding means current
            // stage's 7 have landed; next stage's may still be in flight.
            asm volatile("s_wait_asynccnt 0x7" ::: "memory");
        } else {
            asm volatile("s_wait_asynccnt 0x0" ::: "memory");
        }
        __syncthreads();

        const int ar0 = wm * 32 + l15;
        const int ar1 = ar0 + 16;
        const int k0  = half * 8;
        v16bf ax0 = frag16(&Xs[cur][ar0][k0], &Xs[cur][ar0][k0 + 16]);
        v16bf ax1 = frag16(&Xs[cur][ar1][k0], &Xs[cur][ar1][k0 + 16]);
        v16bf ah0 = frag16(&Hs[cur][ar0][k0], &Hs[cur][ar0][k0 + 16]);
        v16bf ah1 = frag16(&Hs[cur][ar1][k0], &Hs[cur][ar1][k0 + 16]);
        const int brw = wn * 16 + l15;

        {   // r gate: ih + hh sum pre-activation
            v16bf b = frag16(&Ws[cur][0][brw][k0], &Ws[cur][0][brw][k0 + 16]);
            accr[0] = WMMA_BF16(ax0, b, accr[0]);
            accr[1] = WMMA_BF16(ax1, b, accr[1]);
            b = frag16(&Ws[cur][3][brw][k0], &Ws[cur][3][brw][k0 + 16]);
            accr[0] = WMMA_BF16(ah0, b, accr[0]);
            accr[1] = WMMA_BF16(ah1, b, accr[1]);
        }
        {   // z gate
            v16bf b = frag16(&Ws[cur][1][brw][k0], &Ws[cur][1][brw][k0 + 16]);
            accz[0] = WMMA_BF16(ax0, b, accz[0]);
            accz[1] = WMMA_BF16(ax1, b, accz[1]);
            b = frag16(&Ws[cur][4][brw][k0], &Ws[cur][4][brw][k0 + 16]);
            accz[0] = WMMA_BF16(ah0, b, accz[0]);
            accz[1] = WMMA_BF16(ah1, b, accz[1]);
        }
        {   // n gate: i_n and h_n kept separate (r multiplies h_n only)
            v16bf b = frag16(&Ws[cur][2][brw][k0], &Ws[cur][2][brw][k0 + 16]);
            accin[0] = WMMA_BF16(ax0, b, accin[0]);
            accin[1] = WMMA_BF16(ax1, b, accin[1]);
            b = frag16(&Ws[cur][5][brw][k0], &Ws[cur][5][brw][k0 + 16]);
            acchn[0] = WMMA_BF16(ah0, b, acchn[0]);
            acchn[1] = WMMA_BF16(ah1, b, acchn[1]);
        }

        __syncthreads();   // all waves done reading buf[cur] before it is re-staged
        cur = nxt;
    }

    // epilogue: C/D layout -> (vgpr i, lane) = (M = i + 8*half, N = lane&15)
    const int col  = n0 + wn * 16 + l15;
    const float br   = bih[col]             + bhh[col];
    const float bz   = bih[L_DIM + col]     + bhh[L_DIM + col];
    const float bn_i = bih[2 * L_DIM + col];
    const float bn_h = bhh[2 * L_DIM + col];
    #pragma unroll
    for (int s = 0; s < 2; ++s) {
        #pragma unroll
        for (int i = 0; i < 8; ++i) {
            int row  = m0 + wm * 32 + s * 16 + i + half * 8;
            float hp = state[(long)row * (2 * L_DIM) + state_coff + col];
            float r  = sigmoidf(accr[s][i] + br);
            float z  = sigmoidf(accz[s][i] + bz);
            float nn = tanhf(accin[s][i] + bn_i + r * (acchn[s][i] + bn_h));
            float h  = (1.0f - z) * nn + z * hp;
            out[(long)row * (2 * L_DIM) + out_coff + col] = h;
            if (xnext) xnext[(long)row * L_DIM + col] = f2bf_rne(h);
        }
    }
}

extern "C" void kernel_launch(void* const* d_in, const int* in_sizes, int n_in,
                              void* d_out, int out_size, void* d_ws, size_t ws_size,
                              hipStream_t stream) {
    (void)in_sizes; (void)n_in; (void)out_size; (void)ws_size;

    const float* input = (const float*)d_in[0];
    const float* state = (const float*)d_in[1];
    const float* W_ih0 = (const float*)d_in[2];
    const float* W_hh0 = (const float*)d_in[3];
    const float* b_ih0 = (const float*)d_in[4];
    const float* b_hh0 = (const float*)d_in[5];
    const float* W_ih1 = (const float*)d_in[6];
    const float* W_hh1 = (const float*)d_in[7];
    const float* b_ih1 = (const float*)d_in[8];
    const float* b_hh1 = (const float*)d_in[9];
    float* out = (float*)d_out;

    unsigned short* ws = (unsigned short*)d_ws;
    const long NX = (long)B_DIM * L_DIM;        // 16.78M elems
    const long NW = (long)3 * L_DIM * K_DIM;    // 12.58M elems
    unsigned short* X0   = ws;
    unsigned short* H0   = X0 + NX;
    unsigned short* X1   = H0 + NX;
    unsigned short* H1   = X1 + NX;
    unsigned short* Wih0 = H1 + NX;
    unsigned short* Whh0 = Wih0 + NW;
    unsigned short* Wih1 = Whh0 + NW;
    unsigned short* Whh1 = Wih1 + NW;

    dim3 cb(256);
    cvt_f32_bf16<<<4096, cb, 0, stream>>>(input, X0, (int)NX);
    cvt_f32_bf16_strided<<<4096, cb, 0, stream>>>(state, H0, B_DIM, L_DIM, 2 * L_DIM, 0);
    cvt_f32_bf16_strided<<<4096, cb, 0, stream>>>(state, H1, B_DIM, L_DIM, 2 * L_DIM, L_DIM);
    cvt_f32_bf16<<<4096, cb, 0, stream>>>(W_ih0, Wih0, (int)NW);
    cvt_f32_bf16<<<4096, cb, 0, stream>>>(W_hh0, Whh0, (int)NW);
    cvt_f32_bf16<<<4096, cb, 0, stream>>>(W_ih1, Wih1, (int)NW);
    cvt_f32_bf16<<<4096, cb, 0, stream>>>(W_hh1, Whh1, (int)NW);

    dim3 grid(L_DIM / 32, B_DIM / 128);   // 64 x 64 blocks
    gru_layer_wmma<<<grid, 256, 0, stream>>>(X0, H0, Wih0, Whh0, b_ih0, b_hh0,
                                             state, 0, out, 0, X1);
    gru_layer_wmma<<<grid, 256, 0, stream>>>(X1, H1, Wih1, Whh1, b_ih1, b_hh1,
                                             state, L_DIM, out, L_DIM, nullptr);
}
